// LocalFlashK8_3032246911442
// MI455X (gfx1250) — matrix operand order repaired and checked by reading, not compiled, not run on hardware
//
#include <hip/hip_runtime.h>
#include <hip/hip_bf16.h>

// ---------------- types for WMMA ----------------
typedef __bf16 bf16_t;
typedef bf16_t v16bf __attribute__((ext_vector_type(16)));
typedef bf16_t v8bf  __attribute__((ext_vector_type(8)));
typedef float  v8f   __attribute__((ext_vector_type(8)));

#define N_TOK   50000
#define DM      512
#define HEADS   8
#define HD      64
#define KNN     8
#define SCALE_F 0.125f          // 64^-0.5

// LDS A-strip stride (bf16 elements), padded to dodge bank conflicts; 520*2=1040B, 16B aligned
#define LDA 520

// ---------------------------------------------------------------------------
// fp32 -> bf16 bulk convert (4 elems / thread)
// ---------------------------------------------------------------------------
__global__ __launch_bounds__(256) void cvt_f32_bf16_v4(const float* __restrict__ src,
                                                       bf16_t* __restrict__ dst, int n4) {
    int i = blockIdx.x * 256 + threadIdx.x;
    if (i >= n4) return;
    float4 f = ((const float4*)src)[i];
    bf16_t b0 = (bf16_t)f.x, b1 = (bf16_t)f.y, b2 = (bf16_t)f.z, b3 = (bf16_t)f.w;
    unsigned u0 = ((unsigned)__builtin_bit_cast(unsigned short, b1) << 16) |
                   (unsigned)__builtin_bit_cast(unsigned short, b0);
    unsigned u1 = ((unsigned)__builtin_bit_cast(unsigned short, b3) << 16) |
                   (unsigned)__builtin_bit_cast(unsigned short, b2);
    ((uint2*)dst)[i] = make_uint2(u0, u1);
}

// ---------------------------------------------------------------------------
// One wave computes a 16x16 fp32 tile of  A(16xK, LDS bf16) * B(col-rows, global bf16)^T
// A frag: lane l -> row l&15, K at (l>>4)*8 and +16    (ISA 16-bit A 16x32 layout)
// B frag: lane l -> col l&15, K at (l>>4)*8 and +16    (same K order as A)
// ---------------------------------------------------------------------------
__device__ __forceinline__ v8f wave_tile_gemm(const bf16_t* __restrict__ smA,
                                              const bf16_t* __restrict__ B,
                                              int col, int Kdim) {
    const int lane = threadIdx.x & 31;
    const bf16_t* aptr = smA + (lane & 15) * LDA + ((lane >> 4) * 8);
    const bf16_t* bptr = B + (size_t)col * Kdim + ((lane >> 4) * 8);
    v8f acc = {};
    #pragma unroll 4
    for (int kk = 0; kk < Kdim; kk += 32) {
        v8bf a0 = *(const v8bf*)(aptr + kk);
        v8bf a1 = *(const v8bf*)(aptr + kk + 16);
        v8bf b0 = *(const v8bf*)(bptr + kk);
        v8bf b1 = *(const v8bf*)(bptr + kk + 16);
        v16bf av, bv;
        #pragma unroll
        for (int i = 0; i < 8; i++) {
            av[i] = a0[i]; av[i + 8] = a1[i];
            bv[i] = b0[i]; bv[i + 8] = b1[i];
        }
        acc = __builtin_amdgcn_wmma_f32_16x16x32_bf16(false, av, false, bv,
                                                      (short)0, acc, false, false);
    }
    return acc;
}

// stage a 16 x 512 bf16 strip into LDS (padded rows)
__device__ __forceinline__ void stage_A(bf16_t* __restrict__ smA,
                                        const bf16_t* __restrict__ Ab, int r0) {
    const v8bf* src = (const v8bf*)(Ab + (size_t)r0 * DM);   // 1024 16B chunks
    for (int c = threadIdx.x; c < 1024; c += 256) {
        int row = c >> 6;          // 64 chunks per row
        int kc  = (c & 63) * 8;
        *(v8bf*)(smA + row * LDA + kc) = src[c];
    }
}

// ---------------------------------------------------------------------------
// qkv = x @ W_qkv^T   -> split fp32 q/k/v
// ---------------------------------------------------------------------------
__global__ __launch_bounds__(256) void qkv_gemm_kernel(const bf16_t* __restrict__ xb,
                                                       const bf16_t* __restrict__ wb,
                                                       float* __restrict__ q,
                                                       float* __restrict__ k,
                                                       float* __restrict__ v) {
    __shared__ bf16_t smA[16 * LDA];
    const int r0 = blockIdx.x * 16;
    stage_A(smA, xb, r0);
    __syncthreads();

    const int wave = threadIdx.x >> 5;
    const int lane = threadIdx.x & 31;
    const int rowoff = (lane >> 4) * 8;

    for (int ct = wave; ct < 96; ct += 8) {          // 96 column tiles (1536 cols)
        const int col = ct * 16 + (lane & 15);
        v8f acc = wave_tile_gemm(smA, wb, col, DM);
        float* dst; int cc;
        if (col < 512)       { dst = q; cc = col; }
        else if (col < 1024) { dst = k; cc = col - 512; }
        else                 { dst = v; cc = col - 1024; }
        #pragma unroll
        for (int r = 0; r < 8; r++)
            dst[(size_t)(r0 + r + rowoff) * DM + cc] = acc[r];
    }
}

// ---------------------------------------------------------------------------
// gathered K=8 attention; one wave per token; ctx written as bf16
// ---------------------------------------------------------------------------
__global__ __launch_bounds__(256) void attn_kernel(const float* __restrict__ q,
                                                   const float* __restrict__ kf,
                                                   const float* __restrict__ vf,
                                                   const int* __restrict__ idx,
                                                   bf16_t* __restrict__ ctxb) {
    const int wave = threadIdx.x >> 5;
    const int lane = threadIdx.x & 31;
    const int n = blockIdx.x * 8 + wave;
    if (n >= N_TOK) return;

    int js[KNN];
    #pragma unroll
    for (int j = 0; j < KNN; j++) js[j] = idx[n * KNN + j];

    const float2* qp = (const float2*)(q + (size_t)n * DM);

    #pragma unroll 1
    for (int h = 0; h < HEADS; h++) {
        const int d2 = h * 32 + lane;               // float2 index inside row
        float2 qv = qp[d2];
        float s[KNN];
        #pragma unroll
        for (int j = 0; j < KNN; j++) {
            float2 kv = ((const float2*)(kf + (size_t)js[j] * DM))[d2];
            float p = qv.x * kv.x + qv.y * kv.y;
            #pragma unroll
            for (int o = 16; o > 0; o >>= 1) p += __shfl_xor(p, o, 32);
            s[j] = p * SCALE_F;
        }
        float m = s[0];
        #pragma unroll
        for (int j = 1; j < KNN; j++) m = fmaxf(m, s[j]);
        float w[KNN]; float sum = 0.f;
        #pragma unroll
        for (int j = 0; j < KNN; j++) { w[j] = __expf(s[j] - m); sum += w[j]; }
        const float inv = 1.f / sum;
        float2 acc = make_float2(0.f, 0.f);
        #pragma unroll
        for (int j = 0; j < KNN; j++) {
            float2 vv = ((const float2*)(vf + (size_t)js[j] * DM))[d2];
            acc.x += w[j] * inv * vv.x;
            acc.y += w[j] * inv * vv.y;
        }
        bf16_t b0 = (bf16_t)acc.x, b1 = (bf16_t)acc.y;
        unsigned u = ((unsigned)__builtin_bit_cast(unsigned short, b1) << 16) |
                      (unsigned)__builtin_bit_cast(unsigned short, b0);
        ((unsigned*)ctxb)[(size_t)n * 256 + d2] = u;
    }
}

// ---------------------------------------------------------------------------
// out = ctx @ W_out^T + b_out   (fp32 output)
// ---------------------------------------------------------------------------
__global__ __launch_bounds__(256) void out_gemm_kernel(const bf16_t* __restrict__ ctxb,
                                                       const bf16_t* __restrict__ wob,
                                                       const float* __restrict__ bias,
                                                       float* __restrict__ out) {
    __shared__ bf16_t smA[16 * LDA];
    const int r0 = blockIdx.x * 16;
    stage_A(smA, ctxb, r0);
    __syncthreads();

    const int wave = threadIdx.x >> 5;
    const int lane = threadIdx.x & 31;
    const int rowoff = (lane >> 4) * 8;

    for (int ct = wave; ct < 32; ct += 8) {          // 32 column tiles (512 cols)
        const int col = ct * 16 + (lane & 15);
        v8f acc = wave_tile_gemm(smA, wob, col, DM);
        const float b = bias[col];
        #pragma unroll
        for (int r = 0; r < 8; r++)
            out[(size_t)(r0 + r + rowoff) * DM + col] = acc[r] + b;
    }
}

// ---------------------------------------------------------------------------
extern "C" void kernel_launch(void* const* d_in, const int* in_sizes, int n_in,
                              void* d_out, int out_size, void* d_ws, size_t ws_size,
                              hipStream_t stream) {
    const float* x    = (const float*)d_in[0];
    const int*   idx  = (const int*)  d_in[1];
    const float* Wqkv = (const float*)d_in[2];
    const float* Wout = (const float*)d_in[3];
    const float* bout = (const float*)d_in[4];
    float* out = (float*)d_out;

    // workspace carve-up (all offsets 16B aligned); total ~412 MB
    char* ws = (char*)d_ws;
    size_t o = 0;
    bf16_t* xb    = (bf16_t*)(ws + o); o += (size_t)N_TOK * DM * 2;     // 51.2 MB
    bf16_t* wqkvb = (bf16_t*)(ws + o); o += (size_t)3 * DM * DM * 2;    // 1.6 MB
    bf16_t* woutb = (bf16_t*)(ws + o); o += (size_t)DM * DM * 2;        // 0.5 MB
    float*  qf    = (float*)(ws + o);  o += (size_t)N_TOK * DM * 4;     // 102.4 MB
    float*  kf    = (float*)(ws + o);  o += (size_t)N_TOK * DM * 4;
    float*  vf    = (float*)(ws + o);  o += (size_t)N_TOK * DM * 4;
    bf16_t* ctxb  = (bf16_t*)(ws + o); o += (size_t)N_TOK * DM * 2;

    // 1) bf16 conversions
    {
        int n4 = N_TOK * DM / 4;
        cvt_f32_bf16_v4<<<(n4 + 255) / 256, 256, 0, stream>>>(x, xb, n4);
        n4 = 3 * DM * DM / 4;
        cvt_f32_bf16_v4<<<(n4 + 255) / 256, 256, 0, stream>>>(Wqkv, wqkvb, n4);
        n4 = DM * DM / 4;
        cvt_f32_bf16_v4<<<(n4 + 255) / 256, 256, 0, stream>>>(Wout, woutb, n4);
    }

    // 2) QKV projection (WMMA bf16): 3125 row strips
    qkv_gemm_kernel<<<N_TOK / 16, 256, 0, stream>>>(xb, wqkvb, qf, kf, vf);

    // 3) gathered-K attention: 8 tokens per block
    attn_kernel<<<N_TOK / 8, 256, 0, stream>>>(qf, kf, vf, idx, ctxb);

    // 4) output projection + bias (WMMA bf16)
    out_gemm_kernel<<<N_TOK / 16, 256, 0, stream>>>(ctxb, woutb, bout, out);
}
